// MMoE_Model_60000693125188
// MI455X (gfx1250) — matrix operand order repaired and checked by reading, not compiled, not run on hardware
//
#include <hip/hip_runtime.h>
#include <hip/hip_bf16.h>

typedef __attribute__((ext_vector_type(16))) _Float16 v16h;
typedef __attribute__((ext_vector_type(8)))  float    v8f;

#define DEV __device__ __forceinline__

static constexpr int BATCH = 131072;
static constexpr int NBLK  = BATCH / 16;   // 8192 tiles of 16 samples

enum {
  IX_X=0, IX_PW1, IX_PB1, IX_PW2, IX_PB2,
  IX_BW1, IX_BB1, IX_BW2, IX_BB2,
  IX_EW1, IX_EB1, IX_EG1, IX_EBB1, IX_EW2, IX_EB2,
  IX_GW1, IX_GB1, IX_GG1, IX_GBB1, IX_GW2, IX_GB2,
  IX_TW1, IX_TB1, IX_TG1, IX_TBB1,
  IX_TW2, IX_TB2, IX_TG2, IX_TBB2,
  IX_TW3, IX_TB3, IX_TG3, IX_TBB3,
  IX_TW4, IX_TB4, IX_COUNT
};

struct Params {
  const float* in[IX_COUNT];
  float*     out;
  float*     stats;   // ws +0     : 1024 floats (sums / sumsq accumulators)
  float*     bn;      // ws +4096B : 1024 floats (folded BN scale/shift)
  _Float16*  pack;    // ws +8192B : pre-packed WMMA B operands
};

// packed-weight segment bases (in _Float16 units); each 16x16(N)x32(K) block
// is 512 halves = 32 lanes * 16 halves, lane-major (one global_load_b128 x2).
static constexpr int SEG_A = 0;          // exp_W1 : 25 ntiles x 4 ksteps = 100 blocks
static constexpr int SEG_B = 100*512;    // exp_W2 : 5 x (7 x 3)         = 105
static constexpr int SEG_C = 205*512;    // tW1    : 4 x (7 x 4)         = 112
static constexpr int SEG_D = 317*512;    // tW2    : 4 x (4 x 4)         = 64
static constexpr int SEG_E = 381*512;    // tW3    : 4 x (1 x 2)         = 8
static constexpr int SEG_BLOCKS = 389;

DEV float leakyf(float v){ return v >= 0.f ? v : 0.01f*v; }

// ---- WMMA operand builders (contiguous-K order, same in A and B, wave32) -------------------
// A (16x32 f16): lane L row M=L&15 holds K=koff+0..15 contiguous
// with koff=(L>>4)*16  -> element j of v16h maps to K offset koff+j.
DEV v16h load_A_f32(const float* tile, int ld, int kBase) {
  int lane = threadIdx.x;
  const float* row = tile + (lane & 15) * ld + kBase + ((lane >> 4) * 16);
  v16h a;
  #pragma unroll
  for (int j = 0; j < 8; ++j) { a[j] = (_Float16)row[j]; a[8+j] = (_Float16)row[8+j]; }
  return a;
}
DEV v16h load_A_f16(const _Float16* tile, int ld, int kBase) {
  int lane = threadIdx.x;
  const _Float16* row = tile + (lane & 15) * ld + kBase + ((lane >> 4) * 16);
  v16h a;
  #pragma unroll
  for (int j = 0; j < 8; ++j) { a[j] = row[j]; a[8+j] = row[8+j]; }
  return a;
}
DEV v8f wmma_f32A(const float* tile, int ld, const _Float16* bblk, int ksteps) {
  int lane = threadIdx.x;
  v8f c = {};
  for (int kt = 0; kt < ksteps; ++kt) {
    v16h a = load_A_f32(tile, ld, kt * 32);
    v16h b = *reinterpret_cast<const v16h*>(bblk + kt * 512 + lane * 16);
    c = __builtin_amdgcn_wmma_f32_16x16x32_f16(false, a, false, b, (short)0, c, false, false);
  }
  return c;
}
DEV v8f wmma_f16A(const _Float16* tile, int ld, const _Float16* bblk, int ksteps) {
  int lane = threadIdx.x;
  v8f c = {};
  for (int kt = 0; kt < ksteps; ++kt) {
    v16h a = load_A_f16(tile, ld, kt * 32);
    v16h b = *reinterpret_cast<const v16h*>(bblk + kt * 512 + lane * 16);
    c = __builtin_amdgcn_wmma_f32_16x16x32_f16(false, a, false, b, (short)0, c, false, false);
  }
  return c;
}

// ---- scalar front-end: inp = [emb(60) | feat(18) | beta(20) | zeros ->128] --
// Two lanes per sample (part = lane>>4) split the work.
DEV void compute_inp(const Params& P, int base, float* inp_s) {
  int lane = threadIdx.x;
  int m = lane & 15, part = lane >> 4;
  const float* xr = P.in[IX_X] + (size_t)(base + m) * 19;
  // pair nets: 12 pairs, 6 per part
  for (int p = part * 6; p < part * 6 + 6; ++p) {
    float i0 = xr[(p < 6) ? p : (p - 6)];
    float i1 = xr[(p < 6) ? (6 + p) : (12 + p - 6)];
    const float* w1 = P.in[IX_PW1] + p * 40;   // [20][2]
    const float* b1 = P.in[IX_PB1] + p * 20;
    float ph[20];
    #pragma unroll
    for (int j = 0; j < 20; ++j) ph[j] = leakyf(w1[2*j] * i0 + w1[2*j+1] * i1 + b1[j]);
    const float* w2 = P.in[IX_PW2] + p * 100;  // [5][20]
    const float* b2 = P.in[IX_PB2] + p * 5;
    for (int o = 0; o < 5; ++o) {
      float s = b2[o];
      #pragma unroll
      for (int j = 0; j < 20; ++j) s += ph[j] * w2[o*20 + j];
      inp_s[m*128 + p*5 + o] = s;
    }
  }
  // feat passthrough (18), 9 per part
  for (int i = part * 9; i < part * 9 + 9; ++i) inp_s[m*128 + 60 + i] = xr[i];
  // beta net (1 -> 10 -> 20), 10 outputs per part
  float tv = xr[18];
  float hb[10];
  #pragma unroll
  for (int j = 0; j < 10; ++j) hb[j] = leakyf(P.in[IX_BW1][j] * tv + P.in[IX_BB1][j]);
  for (int o = part * 10; o < part * 10 + 10; ++o) {
    float s = P.in[IX_BB2][o];
    #pragma unroll
    for (int j = 0; j < 10; ++j) s += hb[j] * P.in[IX_BW2][o*10 + j];
    inp_s[m*128 + 78 + o] = s;
  }
  // zero pad 98..127 (15 per part)
  for (int i = 98 + part * 15; i < 98 + part * 15 + 15; ++i) inp_s[m*128 + i] = 0.f;
}

// ---- kernel 0: zero the stat accumulators ---------------------------------
__global__ void k_zero(float* stats) {
  for (int i = threadIdx.x; i < 1024; i += blockDim.x) stats[i] = 0.f;
}

// ---- kernel 1: pre-pack weights into WMMA B-operand layout ----------------
// B (32x16 f16): lane L col N=L&15; K = kt*32 + (L>>4)*16 + j for element j.
__global__ __launch_bounds__(32) void k_prepack(Params P) {
  int blk  = blockIdx.x;
  int lane = threadIdx.x;
  int col  = lane & 15;
  int k0   = (lane >> 4) * 16;
  _Float16* dst = P.pack + (size_t)blk * 512 + lane * 16;
  if (blk < 100) {                                     // exp_W1 (5,78,98); N = e*80+o
    int nt = blk / 4, kt = blk % 4;
    int n = nt * 16 + col, e = n / 80, o = n - e * 80;
    int kb = kt * 32 + k0;
    #pragma unroll
    for (int j = 0; j < 16; ++j) {
      int k = kb + j;
      float v = (o < 78 && k < 98) ? P.in[IX_EW1][(e*78 + o)*98 + k] : 0.f;
      dst[j] = (_Float16)v;
    }
  } else if (blk < 205) {                              // exp_W2 (5,100,78)
    int b2 = blk - 100, e = b2 / 21, r = b2 % 21, nt = r / 3, kt = r % 3;
    int n = nt * 16 + col, kb = kt * 32 + k0;
    #pragma unroll
    for (int j = 0; j < 16; ++j) {
      int k = kb + j;
      float v = (n < 100 && k < 78) ? P.in[IX_EW2][(e*100 + n)*78 + k] : 0.f;
      dst[j] = (_Float16)v;
    }
  } else if (blk < 317) {                              // tW1 (4,100,100)
    int b2 = blk - 205, t = b2 / 28, r = b2 % 28, nt = r / 4, kt = r % 4;
    int n = nt * 16 + col, kb = kt * 32 + k0;
    #pragma unroll
    for (int j = 0; j < 16; ++j) {
      int k = kb + j;
      float v = (n < 100 && k < 100) ? P.in[IX_TW1][(t*100 + n)*100 + k] : 0.f;
      dst[j] = (_Float16)v;
    }
  } else if (blk < 381) {                              // tW2 (4,50,100)
    int b2 = blk - 317, t = b2 / 16, r = b2 % 16, nt = r / 4, kt = r % 4;
    int n = nt * 16 + col, kb = kt * 32 + k0;
    #pragma unroll
    for (int j = 0; j < 16; ++j) {
      int k = kb + j;
      float v = (n < 50 && k < 100) ? P.in[IX_TW2][(t*50 + n)*100 + k] : 0.f;
      dst[j] = (_Float16)v;
    }
  } else {                                             // tW3 (4,10,50)
    int b2 = blk - 381, t = b2 / 2, kt = b2 % 2;
    int n = col, kb = kt * 32 + k0;
    #pragma unroll
    for (int j = 0; j < 16; ++j) {
      int k = kb + j;
      float v = (n < 10 && k < 50) ? P.in[IX_TW3][(t*10 + n)*50 + k] : 0.f;
      dst[j] = (_Float16)v;
    }
  }
}

// ---- kernel 2: BN stats pass (he_pre via WMMA, hg_pre scalar) -------------
__global__ __launch_bounds__(32) void k_stats(Params P) {
  __shared__ float inp_s[16*128];
  __shared__ float he_s[16*390];
  __shared__ float hg_s[16*80];
  int lane = threadIdx.x;
  int base = blockIdx.x * 16;
  compute_inp(P, base, inp_s);
  { // hg_pre = feat @ gate_W1^T + gate_b1  (4 tasks x 20 units, K=18)
    int m = lane & 15, part = lane >> 4;
    const float* xr = P.in[IX_X] + (size_t)(base + m) * 19;
    for (int t = part * 2; t < part * 2 + 2; ++t)
      for (int j = 0; j < 20; ++j) {
        float s = P.in[IX_GB1][t*20 + j];
        const float* w = P.in[IX_GW1] + (t*20 + j) * 18;
        #pragma unroll
        for (int i = 0; i < 18; ++i) s += xr[i] * w[i];
        hg_s[m*80 + t*20 + j] = s;
      }
  }
  __syncthreads();
  int col = lane & 15, mb = (lane >> 4) * 8;
  for (int nt = 0; nt < 25; ++nt) {                    // he_pre: K=128, N=400
    v8f c = wmma_f32A(inp_s, 128, P.pack + SEG_A + nt * 4 * 512, 4);
    int n = nt * 16 + col, e = n / 80, o = n - e * 80;
    if (o < 78) {
      float bias = P.in[IX_EB1][e*78 + o];
      #pragma unroll
      for (int r = 0; r < 8; ++r) he_s[(mb + r)*390 + e*78 + o] = c[r] + bias;
    }
  }
  __syncthreads();
  float* heSum = P.stats;       float* heSq = P.stats + 390;
  float* hgSum = P.stats + 780; float* hgSq = P.stats + 860;
  for (int u = lane; u < 390; u += 32) {
    float s1 = 0.f, s2 = 0.f;
    for (int m2 = 0; m2 < 16; ++m2) { float v = he_s[m2*390 + u]; s1 += v; s2 += v*v; }
    atomicAdd(&heSum[u], s1); atomicAdd(&heSq[u], s2);
  }
  for (int u = lane; u < 80; u += 32) {
    float s1 = 0.f, s2 = 0.f;
    for (int m2 = 0; m2 < 16; ++m2) { float v = hg_s[m2*80 + u]; s1 += v; s2 += v*v; }
    atomicAdd(&hgSum[u], s1); atomicAdd(&hgSq[u], s2);
  }
}

// ---- kernel 3: fold BN stats into per-unit scale/shift --------------------
__global__ void k_finalize(Params P) {
  int idx = threadIdx.x;
  const float Binv = 1.0f / (float)BATCH;
  if (idx < 390) {           // experts (stats include bias; z in main excludes it)
    float m = P.stats[idx] * Binv;
    float v = P.stats[390 + idx] * Binv - m * m;
    float sc = P.in[IX_EG1][idx] * rsqrtf(v + 1e-5f);
    P.bn[idx]       = sc;
    P.bn[390 + idx] = (P.in[IX_EB1][idx] - m) * sc + P.in[IX_EBB1][idx];
  } else if (idx < 470) {    // gates (z in main includes bias)
    int u = idx - 390;
    float m = P.stats[780 + u] * Binv;
    float v = P.stats[860 + u] * Binv - m * m;
    float sc = P.in[IX_GG1][u] * rsqrtf(v + 1e-5f);
    P.bn[780 + u] = sc;
    P.bn[860 + u] = -m * sc + P.in[IX_GBB1][u];
  }
}

// row-wise LayerNorm stats over D f16 elements, lanes 0..15 each own a row
DEV void rowstats(const _Float16* z, int stride, int D, float* mu, float* rs) {
  int lane = threadIdx.x;
  if (lane < 16) {
    const _Float16* row = z + lane * stride;
    float s = 0.f;
    for (int i = 0; i < D; ++i) s += (float)row[i];
    float m = s / (float)D;
    float v = 0.f;
    for (int i = 0; i < D; ++i) { float d = (float)row[i] - m; v += d * d; }
    mu[lane] = m; rs[lane] = rsqrtf(v / (float)D + 1e-5f);
  }
}

// ---- kernel 4: main fused pass --------------------------------------------
__global__ __launch_bounds__(32) void k_main(Params P) {
  __shared__ __attribute__((aligned(64))) unsigned char sbuf[56960];
  float*     inp_s   = (float*)sbuf;                    // 2048 f32
  _Float16*  he_s    = (_Float16*)(sbuf + 8192);        // 7680 f16 [16][480] (e*96+o)
  _Float16*  repsA   = (_Float16*)(sbuf + 8192);        // alias: 2048 f16 [16][128]
  _Float16*  z_s     = (_Float16*)(sbuf + 12288);       // alias: 1792 f16
  _Float16*  h1_s    = (_Float16*)(sbuf + 15872);       // alias: 2048 f16 [16][128]
  _Float16*  h2_s    = (_Float16*)(sbuf + 19968);       // alias: 1024 f16 [16][64]
  _Float16*  h3_s    = (_Float16*)(sbuf + 22016);       // alias: 256  f16 [16][16]
  float*     eo_s    = (float*)(sbuf + 23552);          // 8000 f32 [16][5][100]
  float*     gates_s = (float*)(sbuf + 55552);          // 320 f32 [16][4][5]
  float*     mu_s    = (float*)(sbuf + 56832);          // 16
  float*     rs_s    = (float*)(sbuf + 56896);          // 16

  int lane = threadIdx.x;
  int base = blockIdx.x * 16;
  int col = lane & 15, mb = (lane >> 4) * 8;

  compute_inp(P, base, inp_s);

  { // gates: hg_pre (scalar) -> BN fold -> leaky -> gate_W2 -> softmax
    int m = lane & 15, part = lane >> 4;
    const float* xr = P.in[IX_X] + (size_t)(base + m) * 19;
    const float* gsc = P.bn + 780; const float* gsh = P.bn + 860;
    for (int t = part * 2; t < part * 2 + 2; ++t) {
      float hv[20];
      for (int j = 0; j < 20; ++j) {
        float s = P.in[IX_GB1][t*20 + j];
        const float* w = P.in[IX_GW1] + (t*20 + j) * 18;
        #pragma unroll
        for (int i = 0; i < 18; ++i) s += xr[i] * w[i];
        hv[j] = leakyf(s * gsc[t*20 + j] + gsh[t*20 + j]);
      }
      float lg[5]; float mx = -3.0e38f;
      for (int e = 0; e < 5; ++e) {
        float s = P.in[IX_GB2][t*5 + e];
        const float* w = P.in[IX_GW2] + (t*5 + e) * 20;
        #pragma unroll
        for (int j = 0; j < 20; ++j) s += hv[j] * w[j];
        lg[e] = s; mx = fmaxf(mx, s);
      }
      float den = 0.f;
      for (int e = 0; e < 5; ++e) { lg[e] = __expf(lg[e] - mx); den += lg[e]; }
      float inv = 1.f / den;
      for (int e = 0; e < 5; ++e) {
        float g = lg[e] * inv;
        gates_s[m*20 + t*5 + e] = g;
        P.out[(size_t)404 * BATCH + ((size_t)t * BATCH + base + m) * 5 + e] = g;
      }
    }
  }
  // zero he (includes K pads 78..95 per expert)
  for (int i = lane; i < 3840; i += 32) ((unsigned int*)he_s)[i] = 0u;
  __syncthreads();

  // expert layer 1: z = inp@W^T, BN-folded scale/shift + leaky -> he (f16)
  for (int nt = 0; nt < 25; ++nt) {
    v8f c = wmma_f32A(inp_s, 128, P.pack + SEG_A + nt * 4 * 512, 4);
    int n = nt * 16 + col, e = n / 80, o = n - e * 80;
    if (o < 78) {
      float sc = P.bn[e*78 + o], sh = P.bn[390 + e*78 + o];
      #pragma unroll
      for (int r = 0; r < 8; ++r)
        he_s[(mb + r)*480 + e*96 + o] = (_Float16)leakyf(c[r] * sc + sh);
    }
  }
  __syncthreads();

  // expert layer 2: per-expert K=96(pad of 78), N=112(pad of 100)
  for (int e = 0; e < 5; ++e)
    for (int nt = 0; nt < 7; ++nt) {
      v8f c = wmma_f16A(he_s + e * 96, 480, P.pack + SEG_B + ((e*7 + nt) * 3) * 512, 3);
      int n = nt * 16 + col;
      if (n < 100) {
        float bv = P.in[IX_EB2][e*100 + n];
        #pragma unroll
        for (int r = 0; r < 8; ++r) eo_s[((mb + r)*5 + e)*100 + n] = c[r] + bv;
      }
    }
  __syncthreads();

  // towers (per task): reps -> [LN+leaky GEMM] x3 -> scalar head
  for (int t = 0; t < 4; ++t) {
    // reps_t = 0.2 * sum_e gates * eo ; also emitted to d_out
    for (int idx = lane; idx < 2048; idx += 32) {
      int m = idx >> 7, o = idx & 127;
      float v = 0.f;
      if (o < 100) {
        const float* g  = gates_s + m*20 + t*5;
        const float* er = eo_s + m*500 + o;
        v = 0.2f * (g[0]*er[0] + g[1]*er[100] + g[2]*er[200] + g[3]*er[300] + g[4]*er[400]);
        P.out[(size_t)4 * BATCH + (size_t)t * 100 * BATCH + (size_t)(base + m) * 100 + o] = v;
      }
      repsA[idx] = (_Float16)v;
    }
    __syncthreads();
    // layer 1: 100 -> 100
    for (int nt = 0; nt < 7; ++nt) {
      v8f c = wmma_f16A(repsA, 128, P.pack + SEG_C + (t*28 + nt*4) * 512, 4);
      int n = nt * 16 + col;
      if (n < 100) {
        float bv = P.in[IX_TB1][t*100 + n];
        #pragma unroll
        for (int r = 0; r < 8; ++r) z_s[(mb + r)*112 + n] = (_Float16)(c[r] + bv);
      }
    }
    __syncthreads();
    rowstats(z_s, 112, 100, mu_s, rs_s);
    __syncthreads();
    for (int idx = lane; idx < 2048; idx += 32) {
      int m = idx >> 7, o = idx & 127;
      float v = 0.f;
      if (o < 100)
        v = leakyf(((float)z_s[m*112 + o] - mu_s[m]) * rs_s[m] * P.in[IX_TG1][t*100 + o]
                   + P.in[IX_TBB1][t*100 + o]);
      h1_s[idx] = (_Float16)v;
    }
    __syncthreads();
    // layer 2: 100 -> 50
    for (int nt = 0; nt < 4; ++nt) {
      v8f c = wmma_f16A(h1_s, 128, P.pack + SEG_D + (t*16 + nt*4) * 512, 4);
      int n = nt * 16 + col;
      if (n < 50) {
        float bv = P.in[IX_TB2][t*50 + n];
        #pragma unroll
        for (int r = 0; r < 8; ++r) z_s[(mb + r)*64 + n] = (_Float16)(c[r] + bv);
      }
    }
    __syncthreads();
    rowstats(z_s, 64, 50, mu_s, rs_s);
    __syncthreads();
    for (int idx = lane; idx < 1024; idx += 32) {
      int m = idx >> 6, o = idx & 63;
      float v = 0.f;
      if (o < 50)
        v = leakyf(((float)z_s[m*64 + o] - mu_s[m]) * rs_s[m] * P.in[IX_TG2][t*50 + o]
                   + P.in[IX_TBB2][t*50 + o]);
      h2_s[idx] = (_Float16)v;
    }
    __syncthreads();
    // layer 3: 50 -> 10
    {
      v8f c = wmma_f16A(h2_s, 64, P.pack + SEG_E + t * 2 * 512, 2);
      int n = col;
      if (n < 10) {
        float bv = P.in[IX_TB3][t*10 + n];
        #pragma unroll
        for (int r = 0; r < 8; ++r) z_s[(mb + r)*16 + n] = (_Float16)(c[r] + bv);
      }
    }
    __syncthreads();
    rowstats(z_s, 16, 10, mu_s, rs_s);
    __syncthreads();
    for (int idx = lane; idx < 256; idx += 32) {
      int m = idx >> 4, o = idx & 15;
      float v = 0.f;
      if (o < 10)
        v = leakyf(((float)z_s[m*16 + o] - mu_s[m]) * rs_s[m] * P.in[IX_TG3][t*10 + o]
                   + P.in[IX_TBB3][t*10 + o]);
      h3_s[idx] = (_Float16)v;
    }
    __syncthreads();
    // head: 10 -> 1
    if (lane < 16) {
      float s = P.in[IX_TB4][t];
      #pragma unroll
      for (int k = 0; k < 10; ++k) s += (float)h3_s[lane*16 + k] * P.in[IX_TW4][t*10 + k];
      P.out[(size_t)t * BATCH + base + lane] = s;
    }
    __syncthreads();
  }
}

extern "C" void kernel_launch(void* const* d_in, const int* in_sizes, int n_in,
                              void* d_out, int out_size, void* d_ws, size_t ws_size,
                              hipStream_t stream) {
  Params P;
  for (int i = 0; i < IX_COUNT; ++i) P.in[i] = (const float*)d_in[i];
  P.out   = (float*)d_out;
  P.stats = (float*)d_ws;
  P.bn    = P.stats + 1024;
  P.pack  = (_Float16*)((char*)d_ws + 8192);

  k_zero    <<<1, 256, 0, stream>>>(P.stats);
  k_prepack <<<SEG_BLOCKS, 32, 0, stream>>>(P);
  k_stats   <<<NBLK, 32, 0, stream>>>(P);
  k_finalize<<<1, 512, 0, stream>>>(P);
  k_main    <<<NBLK, 32, 0, stream>>>(P);
}
